// PGATClassifier_6176162971771
// MI455X (gfx1250) — matrix operand order repaired and checked by reading, not compiled, not run on hardware
//
#include <hip/hip_runtime.h>
#include <hip/hip_bf16.h>

// ---------------------------------------------------------------------------
// Types for WMMA (gfx1250, wave32)
// ---------------------------------------------------------------------------
typedef __attribute__((ext_vector_type(16))) __bf16 bf16x16;
typedef __attribute__((ext_vector_type(8)))  float  f32x8;
typedef __attribute__((ext_vector_type(4)))  unsigned int uint4v;

#define BM 128
#define BN 128
#define BK 32
#define LDA_S 40   // ushort stride (32 + 8 pad), keeps 16B alignment (80B rows)
#define LDB_S 40

__device__ __forceinline__ unsigned short f2bf(float f) {
    unsigned u = __float_as_uint(f);
    u += 0x7FFFu + ((u >> 16) & 1u);          // round-to-nearest-even
    return (unsigned short)(u >> 16);
}

__device__ __forceinline__ void atomicMaxFloat(float* addr, float val) {
    unsigned* ua = (unsigned*)addr;
    unsigned old = *ua;
    while (__uint_as_float(old) < val) {
        unsigned assumed = old;
        old = atomicCAS(ua, assumed, __float_as_uint(val));
        if (old == assumed) break;
    }
}

// ---------------------------------------------------------------------------
// Tiled GEMM:  C[M,N] = act(A[M,K] @ W[K,N] + bias)
// act: 0 = none, 1 = ELU, 2 = ReLU.   K % 32 == 0, N % 128 == 0 (callers obey)
// 256 threads = 8 waves; wave (wm, wn) computes a 64x32 subtile = 4x2 WMMA tiles
// ---------------------------------------------------------------------------
__global__ __launch_bounds__(256) void gemm_wmma(
    const float* __restrict__ A, int lda,
    const float* __restrict__ W, int ldw,
    const float* __restrict__ bias,
    float* __restrict__ C, int ldc,
    int M, int N, int K, int act)
{
    __shared__ __align__(16) unsigned short Alds[BM * LDA_S];
    __shared__ __align__(16) unsigned short Blds[BN * LDB_S];

    const int tid   = threadIdx.x;
    const int wave  = tid >> 5;
    const int lane  = tid & 31;
    const int waveM = wave >> 2;        // 0..1  -> 64-row group
    const int waveN = wave & 3;         // 0..3  -> 32-col group
    const int blockM = blockIdx.y * BM;
    const int blockN = blockIdx.x * BN;
    const int lrow  = lane & 15;        // M/N index within 16
    const int lhalf = lane >> 4;        // K-half selector per ISA layout

    f32x8 acc[4][2];
#pragma unroll
    for (int mi = 0; mi < 4; ++mi)
#pragma unroll
        for (int ni = 0; ni < 2; ++ni)
            acc[mi][ni] = (f32x8)0.0f;

    for (int k0 = 0; k0 < K; k0 += BK) {
        // ---- stage A tile [BM][BK] f32 -> bf16, row major -------------------
#pragma unroll
        for (int i = 0; i < 4; ++i) {
            int f  = tid + i * 256;           // float4 id, 0..1023
            int r  = f >> 3;                  // row in tile
            int c4 = f & 7;                   // float4 column
            int grow = blockM + r;
            float4 v = make_float4(0.f, 0.f, 0.f, 0.f);
            if (grow < M)
                v = *(const float4*)(A + (size_t)grow * lda + k0 + c4 * 4);
            unsigned p0 = (unsigned)f2bf(v.x) | ((unsigned)f2bf(v.y) << 16);
            unsigned p1 = (unsigned)f2bf(v.z) | ((unsigned)f2bf(v.w) << 16);
            unsigned* dst = (unsigned*)&Alds[r * LDA_S + c4 * 4];
            dst[0] = p0; dst[1] = p1;
        }
        // ---- stage B tile [BK][BN] f32 -> bf16, TRANSPOSED into LDS ---------
#pragma unroll
        for (int i = 0; i < 4; ++i) {
            int f  = tid + i * 256;
            int kr = f >> 5;                  // k row in tile (0..31)
            int c4 = f & 31;                  // float4 column
            int gn = blockN + c4 * 4;
            float4 v = make_float4(0.f, 0.f, 0.f, 0.f);
            if (gn < N)
                v = *(const float4*)(W + (size_t)(k0 + kr) * ldw + gn);
            Blds[(c4 * 4 + 0) * LDB_S + kr] = f2bf(v.x);
            Blds[(c4 * 4 + 1) * LDB_S + kr] = f2bf(v.y);
            Blds[(c4 * 4 + 2) * LDB_S + kr] = f2bf(v.z);
            Blds[(c4 * 4 + 3) * LDB_S + kr] = f2bf(v.w);
        }
        __syncthreads();

        // ---- B fragments (32x16 bf16): lane n = col, k = {0,16} + lhalf*8 + 0..7 ---
        bf16x16 bfrag[2];
#pragma unroll
        for (int ni = 0; ni < 2; ++ni) {
            int n = waveN * 32 + ni * 16 + lrow;
            union { uint4v u[2]; bf16x16 v; } cv;
            cv.u[0] = *(const uint4v*)&Blds[n * LDB_S + lhalf * 8];
            cv.u[1] = *(const uint4v*)&Blds[n * LDB_S + 16 + lhalf * 8];
            bfrag[ni] = cv.v;
        }
        // ---- A fragments (16x32 bf16): k = {0,16} + lhalf*8 + 0..7 ----------
#pragma unroll
        for (int mi = 0; mi < 4; ++mi) {
            int r = waveM * 64 + mi * 16 + lrow;
            union { uint4v u[2]; bf16x16 v; } av;
            av.u[0] = *(const uint4v*)&Alds[r * LDA_S + lhalf * 8];
            av.u[1] = *(const uint4v*)&Alds[r * LDA_S + 16 + lhalf * 8];
            bf16x16 afrag = av.v;
#pragma unroll
            for (int ni = 0; ni < 2; ++ni) {
                acc[mi][ni] = __builtin_amdgcn_wmma_f32_16x16x32_bf16(
                    false, afrag, false, bfrag[ni],
                    (short)0, acc[mi][ni], false, false);
            }
        }
        __syncthreads();
    }

    // ---- epilogue: C layout VGPR r -> M = r + lhalf*8, N = lane&15 ----------
#pragma unroll
    for (int mi = 0; mi < 4; ++mi)
#pragma unroll
        for (int ni = 0; ni < 2; ++ni)
#pragma unroll
            for (int r = 0; r < 8; ++r) {
                int grow = blockM + waveM * 64 + mi * 16 + r + lhalf * 8;
                int gcol = blockN + waveN * 32 + ni * 16 + lrow;
                if (grow < M && gcol < N) {
                    float v = acc[mi][ni][r];
                    if (bias) v += bias[gcol];
                    if (act == 1)      v = v > 0.f ? v : (__expf(v) - 1.0f);
                    else if (act == 2) v = fmaxf(v, 0.0f);
                    C[(size_t)grow * ldc + gcol] = v;
                }
            }
}

// ---------------------------------------------------------------------------
// fill
// ---------------------------------------------------------------------------
__global__ void fill_f32(float* __restrict__ p, float v, size_t n) {
    size_t i = (size_t)blockIdx.x * blockDim.x + threadIdx.x;
    if (i < n) p[i] = v;
}

// ---------------------------------------------------------------------------
// attention scores: es[n,h] = sum_c t[n,h*C+c]*a_src[h,c]  (C == 256 == blockDim)
// ---------------------------------------------------------------------------
__global__ __launch_bounds__(256) void att_scores(
    const float* __restrict__ t, const float* __restrict__ a_src,
    const float* __restrict__ a_dst, float* __restrict__ es,
    float* __restrict__ ed, int H, int C)
{
    __shared__ float red[256];
    const int n = blockIdx.x, tid = threadIdx.x;
    const float* row = t + (size_t)n * H * C;
    for (int h = 0; h < H; ++h) {
        float v = row[h * C + tid];
        float ps = v * a_src[h * C + tid];
        float pd = v * a_dst[h * C + tid];
        red[tid] = ps; __syncthreads();
        for (int s = 128; s > 0; s >>= 1) { if (tid < s) red[tid] += red[tid + s]; __syncthreads(); }
        if (tid == 0) es[n * H + h] = red[0];
        __syncthreads();
        red[tid] = pd; __syncthreads();
        for (int s = 128; s > 0; s >>= 1) { if (tid < s) red[tid] += red[tid + s]; __syncthreads(); }
        if (tid == 0) ed[n * H + h] = red[0];
        __syncthreads();
    }
}

// ---------------------------------------------------------------------------
// edge logits (leaky_relu 0.2) + segment max via float atomic-max
// ---------------------------------------------------------------------------
__global__ void edge_logits(const int* __restrict__ ei, int E, int N, int H,
                            const float* __restrict__ es, const float* __restrict__ ed,
                            float* __restrict__ logits, float* __restrict__ m)
{
    int idx = blockIdx.x * blockDim.x + threadIdx.x;
    int Etot = E + N;
    if (idx >= Etot * H) return;
    int e = idx / H, h = idx % H;
    int s, d;
    if (e < E) { s = ei[e]; d = ei[E + e]; } else { s = d = e - E; }
    float l = es[s * H + h] + ed[d * H + h];
    l = l > 0.f ? l : 0.2f * l;
    logits[idx] = l;
    atomicMaxFloat(&m[d * H + h], l);
}

// exp(logit - m[dst]) -> alpha numerator (in place) + segment sum denominator
__global__ void edge_expsum(const int* __restrict__ ei, int E, int N, int H,
                            float* __restrict__ logits, const float* __restrict__ m,
                            float* __restrict__ den)
{
    int idx = blockIdx.x * blockDim.x + threadIdx.x;
    int Etot = E + N;
    if (idx >= Etot * H) return;
    int e = idx / H, h = idx % H;
    int d = (e < E) ? ei[E + e] : (e - E);
    float ex = __expf(logits[idx] - m[d * H + h]);
    logits[idx] = ex;
    atomicAdd(&den[d * H + h], ex);
}

// out[dst] += t[src] * alpha/(den+eps), one 256-thread block per edge. C == 256
__global__ __launch_bounds__(256) void edge_aggregate(
    const int* __restrict__ ei, int E, int N, int H, int C,
    const float* __restrict__ t, const float* __restrict__ alpha,
    const float* __restrict__ den, float* __restrict__ out)
{
    __shared__ float sc[4];
    const int e = blockIdx.x, tid = threadIdx.x;
    int s, d;
    if (e < E) { s = ei[e]; d = ei[E + e]; } else { s = d = e - E; }
    if (tid < H) sc[tid] = alpha[(size_t)e * H + tid] / (den[d * H + tid] + 1e-16f);
    __syncthreads();
    const int D = H * C;
    const float* srcp = t + (size_t)s * D;
    float* dstp = out + (size_t)d * D;
    __builtin_prefetch(srcp + tid, 0, 1);     // global_prefetch_b8
    for (int f = tid; f < D; f += 256)
        atomicAdd(&dstp[f], srcp[f] * sc[f >> 8]);    // C==256 -> f>>8 = head
}

// ---------------------------------------------------------------------------
// out = elu(layernorm(in + bias) * g + b) (+ residual).  D in {256, 1024}
// ---------------------------------------------------------------------------
__global__ __launch_bounds__(256) void bias_ln_elu(
    const float* __restrict__ in, const float* __restrict__ bias,
    const float* __restrict__ g, const float* __restrict__ b,
    const float* __restrict__ residual, float* __restrict__ out, int D)
{
    __shared__ float red[256];
    const int n = blockIdx.x, tid = threadIdx.x;
    const float* row = in + (size_t)n * D;
    const int per = D >> 8;                   // 1 or 4
    float vals[4];
    float sum = 0.f;
    for (int i = 0; i < per; ++i) {
        int c = tid + i * 256;
        float v = row[c] + bias[c];
        vals[i] = v; sum += v;
    }
    red[tid] = sum; __syncthreads();
    for (int s = 128; s > 0; s >>= 1) { if (tid < s) red[tid] += red[tid + s]; __syncthreads(); }
    float mu = red[0] / (float)D;
    __syncthreads();
    float vs = 0.f;
    for (int i = 0; i < per; ++i) { float dlt = vals[i] - mu; vs += dlt * dlt; }
    red[tid] = vs; __syncthreads();
    for (int s = 128; s > 0; s >>= 1) { if (tid < s) red[tid] += red[tid + s]; __syncthreads(); }
    float rstd = rsqrtf(red[0] / (float)D + 1e-5f);
    __syncthreads();
    float* orow = out + (size_t)n * D;
    for (int i = 0; i < per; ++i) {
        int c = tid + i * 256;
        float v = (vals[i] - mu) * rstd * g[c] + b[c];
        v = v > 0.f ? v : (__expf(v) - 1.0f);
        if (residual) v += residual[(size_t)n * D + c];
        orow[c] = v;
    }
}

// ---------------------------------------------------------------------------
// pooling over graphs (256 features)
// ---------------------------------------------------------------------------
__global__ __launch_bounds__(256) void pool_scatter(
    const float* __restrict__ x3, const int* __restrict__ batch,
    float* __restrict__ psum, float* __restrict__ pmax, float* __restrict__ cnt)
{
    const int n = blockIdx.x, tid = threadIdx.x;
    const int gidx = batch[n];
    float v = x3[(size_t)n * 256 + tid];
    atomicAdd(&psum[gidx * 256 + tid], v);
    atomicMaxFloat(&pmax[gidx * 256 + tid], v);
    if (tid == 0) atomicAdd(&cnt[gidx], 1.0f);
}

__global__ __launch_bounds__(256) void pool_finalize(
    const float* __restrict__ psum, const float* __restrict__ pmax,
    const float* __restrict__ cnt, float* __restrict__ pooled)
{
    const int gidx = blockIdx.x, c = threadIdx.x;
    float s = psum[gidx * 256 + c];
    float m = pmax[gidx * 256 + c];
    float k = fmaxf(cnt[gidx], 1.0f);
    pooled[gidx * 768 + c]       = s / k;
    pooled[gidx * 768 + 256 + c] = m;
    pooled[gidx * 768 + 512 + c] = s;
}

// out[g] = sum_j hid[g,j]*w[j] + b
__global__ __launch_bounds__(128) void cls2_kernel(
    const float* __restrict__ hid, const float* __restrict__ w,
    const float* __restrict__ b, float* __restrict__ out)
{
    __shared__ float red[128];
    const int gidx = blockIdx.x, tid = threadIdx.x;
    red[tid] = hid[gidx * 128 + tid] * w[tid];
    __syncthreads();
    for (int s = 64; s > 0; s >>= 1) { if (tid < s) red[tid] += red[tid + s]; __syncthreads(); }
    if (tid == 0) out[gidx] = red[0] + b[0];
}

// ---------------------------------------------------------------------------
// launch
// ---------------------------------------------------------------------------
extern "C" void kernel_launch(void* const* d_in, const int* in_sizes, int n_in,
                              void* d_out, int out_size, void* d_ws, size_t ws_size,
                              hipStream_t stream) {
    (void)in_sizes; (void)n_in; (void)out_size; (void)ws_size;
    const int N = 20000, E = 320000, G = 32, Etot = E + N;

    const float* x      = (const float*)d_in[0];
    const int*   ei     = (const int*)  d_in[1];
    const int*   batch  = (const int*)  d_in[2];
    const float* proj_w = (const float*)d_in[3];
    const float* proj_b = (const float*)d_in[4];
    const float* gat1_w = (const float*)d_in[5];
    const float* att1_s = (const float*)d_in[6];
    const float* att1_d = (const float*)d_in[7];
    const float* gat1_b = (const float*)d_in[8];
    const float* ln1_g  = (const float*)d_in[9];
    const float* ln1_b  = (const float*)d_in[10];
    const float* gat2_w = (const float*)d_in[11];
    const float* att2_s = (const float*)d_in[12];
    const float* att2_d = (const float*)d_in[13];
    const float* gat2_b = (const float*)d_in[14];
    const float* ln2_g  = (const float*)d_in[15];
    const float* ln2_b  = (const float*)d_in[16];
    const float* gat3_w = (const float*)d_in[17];
    const float* att3_s = (const float*)d_in[18];
    const float* att3_d = (const float*)d_in[19];
    const float* gat3_b = (const float*)d_in[20];
    const float* ln3_g  = (const float*)d_in[21];
    const float* ln3_b  = (const float*)d_in[22];
    const float* cls1_w = (const float*)d_in[23];
    const float* cls1_b = (const float*)d_in[24];
    const float* cls2_w = (const float*)d_in[25];
    const float* cls2_b = (const float*)d_in[26];
    float* out = (float*)d_out;

    float* ws = (float*)d_ws;
    size_t off = 0;
    auto alloc = [&](size_t n) { float* p = ws + off; off += n; return p; };
    float* bufA   = alloc((size_t)N * 1024);   // h0 (ld 768) / t2
    float* bufB   = alloc((size_t)N * 1024);   // t1 / agg2 -> x2
    float* bufC   = alloc((size_t)N * 1024);   // agg1 -> x1
    float* t3     = alloc((size_t)N * 256);
    float* x3     = alloc((size_t)N * 256);
    float* es     = alloc((size_t)N * 4);
    float* ed     = alloc((size_t)N * 4);
    float* mbuf   = alloc((size_t)N * 4);
    float* den    = alloc((size_t)N * 4);
    float* alp    = alloc((size_t)Etot * 4);
    float* cnt    = alloc(G);
    float* psum   = alloc((size_t)G * 256);
    float* pmax   = alloc((size_t)G * 256);
    float* pooled = alloc((size_t)G * 768);
    float* hid    = alloc((size_t)G * 128);

    const dim3 blk(256);
    auto fill = [&](float* p, float v, size_t n) {
        fill_f32<<<dim3((unsigned)((n + 255) / 256)), blk, 0, stream>>>(p, v, n);
    };
    auto gemm = [&](const float* A, int lda, const float* W, int ldw,
                    const float* bias, float* C, int ldc, int M, int Nn, int K, int act) {
        dim3 grid((unsigned)((Nn + BN - 1) / BN), (unsigned)((M + BM - 1) / BM));
        gemm_wmma<<<grid, blk, 0, stream>>>(A, lda, W, ldw, bias, C, ldc, M, Nn, K, act);
    };

    // projection: h0 = elu(x @ proj_w + b)
    gemm(x, 768, proj_w, 768, proj_b, bufA, 768, N, 768, 768, 1);

    // ----- GAT layer 1 (H=4, C=256) -----
    gemm(bufA, 768, gat1_w, 1024, nullptr, bufB, 1024, N, 1024, 768, 0);
    att_scores<<<N, blk, 0, stream>>>(bufB, att1_s, att1_d, es, ed, 4, 256);
    fill(mbuf, -1e30f, (size_t)N * 4);
    fill(den, 0.f, (size_t)N * 4);
    fill(bufC, 0.f, (size_t)N * 1024);
    {
        unsigned nb = (unsigned)((Etot * 4 + 255) / 256);
        edge_logits<<<nb, blk, 0, stream>>>(ei, E, N, 4, es, ed, alp, mbuf);
        edge_expsum<<<nb, blk, 0, stream>>>(ei, E, N, 4, alp, mbuf, den);
        edge_aggregate<<<Etot, blk, 0, stream>>>(ei, E, N, 4, 256, bufB, alp, den, bufC);
    }
    bias_ln_elu<<<N, blk, 0, stream>>>(bufC, gat1_b, ln1_g, ln1_b, nullptr, bufC, 1024);

    // ----- GAT layer 2 (H=4, C=256) + residual -----
    gemm(bufC, 1024, gat2_w, 1024, nullptr, bufA, 1024, N, 1024, 1024, 0);
    att_scores<<<N, blk, 0, stream>>>(bufA, att2_s, att2_d, es, ed, 4, 256);
    fill(mbuf, -1e30f, (size_t)N * 4);
    fill(den, 0.f, (size_t)N * 4);
    fill(bufB, 0.f, (size_t)N * 1024);
    {
        unsigned nb = (unsigned)((Etot * 4 + 255) / 256);
        edge_logits<<<nb, blk, 0, stream>>>(ei, E, N, 4, es, ed, alp, mbuf);
        edge_expsum<<<nb, blk, 0, stream>>>(ei, E, N, 4, alp, mbuf, den);
        edge_aggregate<<<Etot, blk, 0, stream>>>(ei, E, N, 4, 256, bufA, alp, den, bufB);
    }
    bias_ln_elu<<<N, blk, 0, stream>>>(bufB, gat2_b, ln2_g, ln2_b, bufC, bufB, 1024);

    // ----- GAT layer 3 (H=1, C=256, mean == identity) -----
    gemm(bufB, 1024, gat3_w, 256, nullptr, t3, 256, N, 256, 1024, 0);
    att_scores<<<N, blk, 0, stream>>>(t3, att3_s, att3_d, es, ed, 1, 256);
    fill(mbuf, -1e30f, (size_t)N);
    fill(den, 0.f, (size_t)N);
    fill(x3, 0.f, (size_t)N * 256);
    {
        unsigned nb = (unsigned)((Etot + 255) / 256);
        edge_logits<<<nb, blk, 0, stream>>>(ei, E, N, 1, es, ed, alp, mbuf);
        edge_expsum<<<nb, blk, 0, stream>>>(ei, E, N, 1, alp, mbuf, den);
        edge_aggregate<<<Etot, blk, 0, stream>>>(ei, E, N, 1, 256, t3, alp, den, x3);
    }
    bias_ln_elu<<<N, blk, 0, stream>>>(x3, gat3_b, ln3_g, ln3_b, nullptr, x3, 256);

    // ----- pooling: [mean | max | sum] -> [G, 768] -----
    fill(cnt, 0.f, G);
    fill(psum, 0.f, (size_t)G * 256);
    fill(pmax, -1e30f, (size_t)G * 256);
    pool_scatter<<<N, blk, 0, stream>>>(x3, batch, psum, pmax, cnt);
    pool_finalize<<<G, blk, 0, stream>>>(psum, pmax, cnt, pooled);

    // ----- classifier -----
    gemm(pooled, 768, cls1_w, 128, cls1_b, hid, 128, G, 128, 768, 2);  // ReLU
    cls2_kernel<<<G, dim3(128), 0, stream>>>(hid, cls2_w, cls2_b, out);
}
